// Feat_Expansion_Attn_66331474920198
// MI455X (gfx1250) — matrix operand order repaired and checked by reading, not compiled, not run on hardware
//
#include <hip/hip_runtime.h>
#include <hip/hip_fp16.h>

typedef _Float16 h16;
typedef __attribute__((ext_vector_type(16))) _Float16 v16h;
typedef __attribute__((ext_vector_type(8)))  _Float16 v8h;
typedef __attribute__((ext_vector_type(8)))  float    v8f;
typedef __attribute__((ext_vector_type(4)))  int      v4i;

#define DMODEL 512
#define NQ     800
#define NKV    1600
#define BATCH  8
#define HEADS  8
#define HD     64

// ---------------------------------------------------------------------------
// Async global->LDS tile copy (CDNA5 GLOBAL_LOAD_ASYNC_TO_LDS_B128, ASYNCcnt),
// guarded so compilation never regresses on toolchains without the builtin.
// Builtin signature (from hipcc diagnostic): (int4 AS1*, int4 AS3*, imm, imm).
// ---------------------------------------------------------------------------
#if __has_builtin(__builtin_amdgcn_global_load_async_to_lds_b128) && \
    __has_builtin(__builtin_amdgcn_s_wait_asynccnt)
#define ASYNC_COPY 1
typedef __attribute__((address_space(1))) v4i g_v4i;
typedef __attribute__((address_space(3))) v4i l_v4i;
#else
#define ASYNC_COPY 0
#endif

__device__ __forceinline__ void cp128(const h16* __restrict__ g, h16* l)
{
#if ASYNC_COPY
    __builtin_amdgcn_global_load_async_to_lds_b128(
        (g_v4i*)g, (l_v4i*)l, 0, 0);
#else
    *(v8h*)l = *(const v8h*)g;
#endif
}

__device__ __forceinline__ void cp_wait()
{
#if ASYNC_COPY
    __builtin_amdgcn_s_wait_asynccnt(0);
#endif
}

// ---------------------------------------------------------------------------
// Tiled GEMM:  out[M,512] = A[M,512](f16) @ W[512,512]  (W given pre-transposed
// as Wt[n][k], f16).  128 threads = 4 waves; block tile 128x64; K-chunk 32;
// double-buffered LDS with async staging. Each wave: 8 WMMAs / chunk.
// ---------------------------------------------------------------------------
__global__ __launch_bounds__(128)
void gemm_d512(const h16* __restrict__ A, const h16* __restrict__ Wt,
               const float* __restrict__ bias, float scale, int relu,
               float* __restrict__ outF, h16* __restrict__ outH)
{
    __shared__ __align__(32) h16 lsa[2][128 * 32];   // [m][k]
    __shared__ __align__(32) h16 lsb[2][64 * 32];    // [n][k]
    const int tid  = threadIdx.x;
    const int lane = tid & 31;
    const int wave = tid >> 5;
    const int g    = lane >> 4;
    const int ln   = lane & 15;
    const int m0 = blockIdx.x * 128;
    const int n0 = blockIdx.y * 64;

    auto stage = [&](int buf, int k0) {
#pragma unroll
        for (int u = 0; u < 4; ++u) {                 // A tile: 512 b128 units
            int e = tid + u * 128;
            int row = e >> 2, kc = (e & 3) * 8;
            cp128(A + (size_t)(m0 + row) * DMODEL + k0 + kc, &lsa[buf][row * 32 + kc]);
        }
#pragma unroll
        for (int u = 0; u < 2; ++u) {                 // B tile: 256 b128 units
            int e = tid + u * 128;
            int n = e >> 2, kc = (e & 3) * 8;
            cp128(Wt + (size_t)(n0 + n) * DMODEL + k0 + kc, &lsb[buf][n * 32 + kc]);
        }
    };

    v8f acc[2][4] = {};
    stage(0, 0);
    for (int kc = 0; kc < 16; ++kc) {
        const int buf = kc & 1;
        cp_wait();
        __syncthreads();
        if (kc + 1 < 16) stage(buf ^ 1, (kc + 1) * 32);

        // load ALL fragments first, then issue the 8 WMMAs back-to-back
        v16h af[2], bf[4];
#pragma unroll
        for (int s = 0; s < 2; ++s) {
            const h16* ap = &lsa[buf][(wave * 32 + s * 16 + ln) * 32];
            v8h lo = *(const v8h*)(ap + g * 16);
            v8h hi = *(const v8h*)(ap + g * 16 + 8);
#pragma unroll
            for (int i = 0; i < 8; ++i) { af[s][i] = lo[i]; af[s][i + 8] = hi[i]; }
        }
#pragma unroll
        for (int ng = 0; ng < 4; ++ng)
            bf[ng] = *(const v16h*)(&lsb[buf][(ng * 16 + ln) * 32 + g * 16]);

#pragma unroll
        for (int s = 0; s < 2; ++s)
#pragma unroll
            for (int ng = 0; ng < 4; ++ng)
                acc[s][ng] = __builtin_amdgcn_wmma_f32_16x16x32_f16(
                    false, af[s], false, bf[ng], (short)0, acc[s][ng], false, false);
    }

#pragma unroll
    for (int s = 0; s < 2; ++s)
#pragma unroll
        for (int ng = 0; ng < 4; ++ng) {
            int col = n0 + ng * 16 + ln;
            float bv = bias ? bias[col] : 0.f;
#pragma unroll
            for (int r = 0; r < 8; ++r) {
                int row = m0 + wave * 32 + s * 16 + r + 8 * g;
                float v = acc[s][ng][r] * scale + bv;
                if (relu) v = fmaxf(v, 0.f);
                if (outF) outF[(size_t)row * DMODEL + col] = v;
                if (outH) outH[(size_t)row * DMODEL + col] = (h16)v;
            }
        }
}

// ---------------------------------------------------------------------------
// Flash attention: 1 wave per (qtile, head, batch). Q pre-scaled by 1/sqrt(hd).
// ---------------------------------------------------------------------------
__global__ __launch_bounds__(32)
void flash_attn(const h16* __restrict__ Q, const h16* __restrict__ K,
                const h16* __restrict__ V, h16* __restrict__ O, int Nk)
{
    __shared__ __align__(32) h16 lq[16 * 64];   // [m][d]
    __shared__ __align__(32) h16 lk[32 * 64];   // [n][d]
    __shared__ __align__(32) h16 lv[64 * 32];   // [d][n]
    __shared__ __align__(32) h16 lp[16 * 32];   // [m][n]

    const int lane = threadIdx.x;
    const int g = lane >> 4, ln = lane & 15;
    const int qt = blockIdx.x, hh = blockIdx.y, b = blockIdx.z;
    const size_t qbase  = ((size_t)b * NQ + qt * 16) * DMODEL + hh * HD;
    const size_t kvbase = (size_t)b * Nk * DMODEL + hh * HD;

    for (int u = lane; u < 16 * 8; u += 32) {
        int row = u >> 3, kc = (u & 7) * 8;
        *(v8h*)(lq + row * 64 + kc) = *(const v8h*)(Q + qbase + (size_t)row * DMODEL + kc);
    }
    __syncthreads();

    v16h aq[2];
#pragma unroll
    for (int dh = 0; dh < 2; ++dh) {
        const h16* ap = lq + ln * 64 + dh * 32;
        v8h lo = *(const v8h*)(ap + g * 16);
        v8h hi = *(const v8h*)(ap + g * 16 + 8);
#pragma unroll
        for (int i = 0; i < 8; ++i) { aq[dh][i] = lo[i]; aq[dh][i + 8] = hi[i]; }
    }

    float rm[8], rl[8];
#pragma unroll
    for (int r = 0; r < 8; ++r) { rm[r] = -1e30f; rl[r] = 0.f; }
    v8f o0 = {}, o1 = {}, o2 = {}, o3 = {};

    for (int k0 = 0; k0 < Nk; k0 += 32) {
        __syncthreads();
        for (int u = lane; u < 32 * 8; u += 32) {                 // K chunk 32x64
            int row = u >> 3, kc = (u & 7) * 8;
            *(v8h*)(lk + row * 64 + kc) =
                *(const v8h*)(K + kvbase + (size_t)(k0 + row) * DMODEL + kc);
        }
        {                                                         // V chunk transposed
            const h16* vp = V + kvbase + (size_t)(k0 + lane) * DMODEL;
#pragma unroll
            for (int dc = 0; dc < 8; ++dc) {
                v8h v = *(const v8h*)(vp + dc * 8);
#pragma unroll
                for (int i = 0; i < 8; ++i) lv[(dc * 8 + i) * 32 + lane] = v[i];
            }
        }
        __syncthreads();

        v8f s0 = {}, s1 = {};
#pragma unroll
        for (int dh = 0; dh < 2; ++dh) {
            v16h b0 = *(const v16h*)(lk + ln * 64 + dh * 32 + g * 16);
            v16h b1 = *(const v16h*)(lk + (16 + ln) * 64 + dh * 32 + g * 16);
            s0 = __builtin_amdgcn_wmma_f32_16x16x32_f16(false, aq[dh], false, b0, (short)0, s0, false, false);
            s1 = __builtin_amdgcn_wmma_f32_16x16x32_f16(false, aq[dh], false, b1, (short)0, s1, false, false);
        }

#pragma unroll
        for (int r = 0; r < 8; ++r) {
            float mx = fmaxf(s0[r], s1[r]);
#pragma unroll
            for (int off = 1; off < 16; off <<= 1)
                mx = fmaxf(mx, __shfl_xor(mx, off, 32));
            float mnew = fmaxf(rm[r], mx);
            float sc = __expf(rm[r] - mnew);
            rm[r] = mnew;
            float e0 = __expf(s0[r] - mnew);
            float e1 = __expf(s1[r] - mnew);
            s0[r] = e0; s1[r] = e1;
            float rs = e0 + e1;
#pragma unroll
            for (int off = 1; off < 16; off <<= 1)
                rs += __shfl_xor(rs, off, 32);
            rl[r] = rl[r] * sc + rs;
            o0[r] *= sc; o1[r] *= sc; o2[r] *= sc; o3[r] *= sc;
        }

#pragma unroll
        for (int r = 0; r < 8; ++r) {
            lp[(r + 8 * g) * 32 + ln]      = (h16)s0[r];
            lp[(r + 8 * g) * 32 + 16 + ln] = (h16)s1[r];
        }
        __syncthreads();
        v16h apf;
        {
            const h16* pp = lp + ln * 32;
            v8h lo = *(const v8h*)(pp + g * 16);
            v8h hi = *(const v8h*)(pp + g * 16 + 8);
#pragma unroll
            for (int i = 0; i < 8; ++i) { apf[i] = lo[i]; apf[i + 8] = hi[i]; }
        }
        v16h bv0 = *(const v16h*)(lv + (0 * 16 + ln) * 32 + g * 16);
        v16h bv1 = *(const v16h*)(lv + (1 * 16 + ln) * 32 + g * 16);
        v16h bv2 = *(const v16h*)(lv + (2 * 16 + ln) * 32 + g * 16);
        v16h bv3 = *(const v16h*)(lv + (3 * 16 + ln) * 32 + g * 16);
        o0 = __builtin_amdgcn_wmma_f32_16x16x32_f16(false, apf, false, bv0, (short)0, o0, false, false);
        o1 = __builtin_amdgcn_wmma_f32_16x16x32_f16(false, apf, false, bv1, (short)0, o1, false, false);
        o2 = __builtin_amdgcn_wmma_f32_16x16x32_f16(false, apf, false, bv2, (short)0, o2, false, false);
        o3 = __builtin_amdgcn_wmma_f32_16x16x32_f16(false, apf, false, bv3, (short)0, o3, false, false);
    }

#pragma unroll
    for (int r = 0; r < 8; ++r) {
        float inv = 1.0f / rl[r];
        int row = qt * 16 + r + 8 * g;
        size_t base = ((size_t)b * NQ + row) * DMODEL + hh * HD;
        O[base +  0 + ln] = (h16)(o0[r] * inv);
        O[base + 16 + ln] = (h16)(o1[r] * inv);
        O[base + 32 + ln] = (h16)(o2[r] * inv);
        O[base + 48 + ln] = (h16)(o3[r] * inv);
    }
}

// ---------------------------------------------------------------------------
// Fused residual add + LayerNorm (1 wave per 512-wide token row)
// ---------------------------------------------------------------------------
__global__ __launch_bounds__(128)
void add_ln(float* __restrict__ qcur, const float* __restrict__ delta,
            const float* __restrict__ gamma, const float* __restrict__ beta,
            h16* __restrict__ qh, int Ntok)
{
    int wave = threadIdx.x >> 5, lane = threadIdx.x & 31;
    int tok = blockIdx.x * 4 + wave;
    if (tok >= Ntok) return;
    float* row = qcur + (size_t)tok * DMODEL;
    const float* drow = delta + (size_t)tok * DMODEL;
    float v[16];
    float s = 0.f;
#pragma unroll
    for (int i = 0; i < 16; ++i) {
        int c = lane + i * 32;
        v[i] = row[c] + drow[c];
        s += v[i];
    }
#pragma unroll
    for (int off = 1; off < 32; off <<= 1) s += __shfl_xor(s, off, 32);
    float mean = s * (1.0f / DMODEL);
    float var = 0.f;
#pragma unroll
    for (int i = 0; i < 16; ++i) { float d = v[i] - mean; var += d * d; }
#pragma unroll
    for (int off = 1; off < 32; off <<= 1) var += __shfl_xor(var, off, 32);
    float rstd = rsqrtf(var * (1.0f / DMODEL) + 1e-5f);
#pragma unroll
    for (int i = 0; i < 16; ++i) {
        int c = lane + i * 32;
        float o = (v[i] - mean) * rstd * gamma[c] + beta[c];
        row[c] = o;
        qh[(size_t)tok * DMODEL + c] = (h16)o;
    }
}

// ---------------------------------------------------------------------------
// utility kernels
// ---------------------------------------------------------------------------
// W[k][n] f32 -> Wt[n][k] f16, LDS-tiled transpose (one-time per launch)
__global__ __launch_bounds__(256)
void transpose_w_k(const float* __restrict__ W, h16* __restrict__ Wt)
{
    __shared__ float tile[32][33];
    int bk = blockIdx.x * 32, bn = blockIdx.y * 32;
    int tx = threadIdx.x & 31, ty = threadIdx.x >> 5;     // ty: 0..7
    for (int i = ty; i < 32; i += 8)
        tile[i][tx] = W[(size_t)(bk + i) * DMODEL + bn + tx];
    __syncthreads();
    for (int i = ty; i < 32; i += 8)
        Wt[(size_t)(bn + i) * DMODEL + bk + tx] = (h16)tile[tx][i];
}

__global__ void x_to_mem_k(const float* __restrict__ x, h16* __restrict__ memh)
{
    int c = blockIdx.x, b = blockIdx.y;
    const float* src = x + ((size_t)b * DMODEL + c) * 1600;
    h16* dst = memh + (size_t)b * NKV * DMODEL + c;
    for (int n = threadIdx.x; n < 1600; n += blockDim.x)
        dst[(size_t)n * DMODEL] = (h16)src[n];
}

__global__ void init_q_k(const float* __restrict__ qe, float* __restrict__ qc,
                         h16* __restrict__ qh)
{
    size_t i = (size_t)blockIdx.x * blockDim.x + threadIdx.x;
    if (i >= (size_t)NQ * DMODEL) return;
    float v = qe[i];
    for (int b = 0; b < BATCH; ++b) {
        qc[(size_t)b * NQ * DMODEL + i] = v;
        qh[(size_t)b * NQ * DMODEL + i] = (h16)v;
    }
}

__global__ void write_center_k(const float* __restrict__ x, float* __restrict__ out)
{
    int bc = blockIdx.x;                       // b*512 + c
    const float* src = x + (size_t)bc * 1600;
    float* dst = out + (size_t)bc * 14400;
    for (int t = threadIdx.x; t < 1600; t += blockDim.x) {
        int i = t / 40, j = t % 40;
        dst[(40 + i) * 120 + (40 + j)] = src[t];
    }
}

__global__ void scatter_ring_k(const float* __restrict__ q, const int* __restrict__ coords,
                               float* __restrict__ out)
{
    int t = blockIdx.x * blockDim.x + threadIdx.x;   // 0..12799
    int b = blockIdx.y;
    if (t >= 12800) return;
    int ci = coords[2 * t], cj = coords[2 * t + 1];
    const float* qrow = q + ((size_t)b * NQ + (t >> 4)) * DMODEL;
    float* obase = out + (size_t)b * DMODEL * 14400 + ci * 120 + cj;
    for (int c = 0; c < DMODEL; ++c)
        obase[(size_t)c * 14400] = qrow[c];
}

// ---------------------------------------------------------------------------
extern "C" void kernel_launch(void* const* d_in, const int* in_sizes, int n_in,
                              void* d_out, int out_size, void* d_ws, size_t ws_size,
                              hipStream_t stream)
{
    const float* x    = (const float*)d_in[0];
    const float* qe   = (const float*)d_in[1];
    const float* Wf[10] = {
        (const float*)d_in[2],  (const float*)d_in[3],  (const float*)d_in[4],
        (const float*)d_in[5],  (const float*)d_in[8],  (const float*)d_in[9],
        (const float*)d_in[10], (const float*)d_in[11], (const float*)d_in[14],
        (const float*)d_in[16] };
    const float* ln1g = (const float*)d_in[6];
    const float* ln1b = (const float*)d_in[7];
    const float* ln2g = (const float*)d_in[12];
    const float* ln2b = (const float*)d_in[13];
    const float* b1   = (const float*)d_in[15];
    const float* b2   = (const float*)d_in[17];
    const float* ln3g = (const float*)d_in[18];
    const float* ln3b = (const float*)d_in[19];
    const int*  coords = (const int*)d_in[20];
    float* out = (float*)d_out;

    char* ws = (char*)d_ws;
    size_t off = 0;
    auto alloc = [&](size_t bytes) -> void* {
        void* p = ws + off;
        off = (off + bytes + 255) & ~(size_t)255;
        return p;
    };

    const size_t WELEM = (size_t)DMODEL * DMODEL;       // 262144
    h16* Wt[10];
    for (int i = 0; i < 10; ++i) Wt[i] = (h16*)alloc(WELEM * 2);
    h16*   memh = (h16*)  alloc((size_t)BATCH * NKV * DMODEL * 2);
    float* qcur = (float*)alloc((size_t)BATCH * NQ * DMODEL * 4);
    h16*   qh   = (h16*)  alloc((size_t)BATCH * NQ * DMODEL * 2);
    h16*   qp   = (h16*)  alloc((size_t)BATCH * NQ * DMODEL * 2);
    h16*   kh   = (h16*)  alloc((size_t)BATCH * NKV * DMODEL * 2);
    h16*   vh   = (h16*)  alloc((size_t)BATCH * NKV * DMODEL * 2);
    h16*   oh   = (h16*)  alloc((size_t)BATCH * NQ * DMODEL * 2);
    float* delta= (float*)alloc((size_t)BATCH * NQ * DMODEL * 4);
    h16*   hid  = (h16*)  alloc((size_t)BATCH * NQ * DMODEL * 2);

    const int MQ = BATCH * NQ;      // 6400 token rows
    const int MM = BATCH * NKV;     // 12800 memory rows
    const float qscale = 0.125f;    // 1/sqrt(64)

    // --- setup: transpose weights (f16), build mem tokens, broadcast queries
    for (int i = 0; i < 10; ++i)
        transpose_w_k<<<dim3(16, 16), 256, 0, stream>>>(Wf[i], Wt[i]);
    x_to_mem_k<<<dim3(DMODEL, BATCH), 256, 0, stream>>>(x, memh);
    init_q_k<<<(NQ * DMODEL + 255) / 256, 256, 0, stream>>>(qe, qcur, qh);

    auto gemm = [&](const h16* A, int wi, const float* bias, float scale, int relu,
                    float* oF, h16* oH, int M) {
        gemm_d512<<<dim3(M / 128, DMODEL / 64), 128, 0, stream>>>(
            A, Wt[wi], bias, scale, relu, oF, oH);
    };

    // --- self attention ---
    gemm(qh, 0, nullptr, qscale, 0, nullptr, qp, MQ);      // Q (pre-scaled)
    gemm(qh, 1, nullptr, 1.f,    0, nullptr, kh, MQ);      // K
    gemm(qh, 2, nullptr, 1.f,    0, nullptr, vh, MQ);      // V
    flash_attn<<<dim3(NQ / 16, HEADS, BATCH), 32, 0, stream>>>(qp, kh, vh, oh, NQ);
    gemm(oh, 3, nullptr, 1.f,    0, delta, nullptr, MQ);   // O proj
    add_ln<<<(MQ + 3) / 4, 128, 0, stream>>>(qcur, delta, ln1g, ln1b, qh, MQ);

    // --- cross attention ---
    gemm(qh,   4, nullptr, qscale, 0, nullptr, qp, MQ);
    gemm(memh, 5, nullptr, 1.f,    0, nullptr, kh, MM);
    gemm(memh, 6, nullptr, 1.f,    0, nullptr, vh, MM);
    flash_attn<<<dim3(NQ / 16, HEADS, BATCH), 32, 0, stream>>>(qp, kh, vh, oh, NKV);
    gemm(oh, 7, nullptr, 1.f, 0, delta, nullptr, MQ);
    add_ln<<<(MQ + 3) / 4, 128, 0, stream>>>(qcur, delta, ln2g, ln2b, qh, MQ);

    // --- FFN ---
    gemm(qh,  8, b1, 1.f, 1, nullptr, hid, MQ);            // relu(q@W1+b1)
    gemm(hid, 9, b2, 1.f, 0, delta, nullptr, MQ);
    add_ln<<<(MQ + 3) / 4, 128, 0, stream>>>(qcur, delta, ln3g, ln3b, qh, MQ);

    // --- output assembly ---
    write_center_k<<<BATCH * DMODEL, 256, 0, stream>>>(x, out);
    scatter_ring_k<<<dim3(50, BATCH), 256, 0, stream>>>(qcur, coords, out);
}
